// message_mlp_15633680958323
// MI455X (gfx1250) — matrix operand order repaired and checked by reading, not compiled, not run on hardware
//
#include <hip/hip_runtime.h>
#include <stdint.h>

// ---- problem constants ----
#define Bn   16
#define Mn   512
#define Tn   64
#define Fn   128
#define Hn   128

typedef __attribute__((ext_vector_type(16))) _Float16 v16h;
typedef __attribute__((ext_vector_type(8)))  float    v8f;
typedef __attribute__((ext_vector_type(4)))  unsigned int u32x4;
typedef __attribute__((ext_vector_type(8)))  int      i32x8;
typedef __attribute__((ext_vector_type(4)))  int      i32x4;

// -------------------------------------------------------------------------
// Tensor Data Mover: async 2D tile load (16 rows x 128 f32) global -> LDS.
// D# per CDNA5 ISA (group0: count/lds/global addr/type,
// group1: data_size=4B, dims/strides/tile dims). 2D tensor -> groups 2/3 zero.
// 6-arg builtin on this toolchain. The "memory" clobber tells the optimizer
// that LDS is written here, so later (non-volatile) reads of the destination
// buffer are kept and may still be vectorized into ds_load_b128.
// -------------------------------------------------------------------------
__device__ inline void issue_tdm_tile(const float* gsrc, void* ldsdst) {
  unsigned long long ga = (unsigned long long)(uintptr_t)gsrc;
  unsigned int lds = (unsigned int)(uintptr_t)ldsdst;   // low 32 bits = LDS byte addr
  u32x4 g0;
  g0[0] = 1u;                                            // count = 1 valid descriptor
  g0[1] = lds;                                           // lds_addr
  g0[2] = (unsigned int)ga;                              // global_addr[31:0]
  g0[3] = (unsigned int)((ga >> 32) & 0x1FFFFFFull) | (2u << 30); // addr[56:32] | type=2
  i32x8 g1;
  g1[0] = (int)(2u << 16);          // workgroup_mask=0, data_size=2 (4 bytes)
  g1[1] = (int)(128u << 16);        // tensor_dim0 = 128 (bits 79:64 part)
  g1[2] = (int)(64u  << 16);        // tensor_dim0 hi=0 | tensor_dim1 = 64
  g1[3] = (int)(128u << 16);        // tensor_dim1 hi=0 | tile_dim0 = 128
  g1[4] = 16;                       // tile_dim1 = 16, tile_dim2 = 0
  g1[5] = 128;                      // tensor_dim0_stride = 128 elements
  g1[6] = 0;
  g1[7] = 0;
  i32x4 z4 = {0, 0, 0, 0};
  i32x8 z8 = {0, 0, 0, 0, 0, 0, 0, 0};
  __builtin_amdgcn_tensor_load_to_lds(g0, g1, z4, z4, z8, 0);
  asm volatile("" ::: "memory");
}

__device__ inline void wait_tdm() {
  __builtin_amdgcn_s_wait_tensorcnt(0);
  asm volatile("" ::: "memory");
}

// -------------------------------------------------------------------------
// prep kernels
// -------------------------------------------------------------------------
__global__ void zero_kernel(float* __restrict__ p, int n) {
  int i = blockIdx.x * blockDim.x + threadIdx.x;
  if (i < n) p[i] = 0.0f;
}

// Pack a row-major [Nout][ld] f32 matrix (cols col_off..col_off+K) into f16
// WMMA B fragments: frag=(ntile,kfrag) of 32 lanes x 16 halves,
//   n = ntile*16 + lane%16,  k = kfrag*32 + (i<8 ? i : i+8) + 8*(lane/16)
__global__ void pack_b_kernel(const float* __restrict__ src, int ld, int col_off,
                              _Float16* __restrict__ dst, int total) {
  int idx = blockIdx.x * blockDim.x + threadIdx.x;
  if (idx >= total) return;
  int i    = idx & 15;
  int lane = (idx >> 4) & 31;
  int frag = idx >> 9;
  int kf   = frag & 3;
  int nt   = frag >> 2;
  int n = nt * 16 + (lane & 15);
  int k = kf * 32 + ((i < 8) ? i : i + 8) + 8 * (lane >> 4);
  dst[idx] = (_Float16)src[n * ld + col_off + k];
}

// g_target[b,t,h] = h_target[b,t,:] . mlp_w1[h, 0:128] + mlp_b1[h]
__global__ void gtarget_kernel(const float* __restrict__ ht, const float* __restrict__ w1,
                               const float* __restrict__ b1, float* __restrict__ gt) {
  int bt = blockIdx.x;          // b*T + t
  int h  = threadIdx.x;         // 0..127
  __shared__ float row[Hn];
  row[h] = ht[bt * Hn + h];
  __syncthreads();
  float s = b1[h];
  const float* wrow = w1 + h * (2 * Hn);
  for (int f = 0; f < Hn; ++f) s += row[f] * wrow[f];
  gt[bt * Hn + h] = s;
}

// -------------------------------------------------------------------------
// Fused message kernel.
// grid = (mchunk 0..7, ttile 0..3, b 0..15); 256 threads = 8 waves.
// Wave w owns output columns [16w, 16w+16) of H. Loops over 64 messages,
// TDM double-buffers X tiles into LDS.
// -------------------------------------------------------------------------
__global__ __launch_bounds__(256)
void fused_msg_kernel(const float* __restrict__ X, const float* __restrict__ Cc,
                      const float* __restrict__ dist,
                      const _Float16* __restrict__ linPack,
                      const _Float16* __restrict__ w1Pack,
                      const float* __restrict__ lin_b, const float* __restrict__ w2,
                      const float* __restrict__ b2p, const float* __restrict__ l2w,
                      const float* __restrict__ l2b, const float* __restrict__ gt,
                      float* __restrict__ hagg) {
  const int mchunk = blockIdx.x;
  const int tt = blockIdx.y;
  const int b  = blockIdx.z;
  const int t0 = tt * 16;
  const int tid  = threadIdx.x;
  const int w    = tid >> 5;
  const int lane = tid & 31;
  const int col  = (w << 4) + (lane & 15);   // H column this lane owns in C layout
  const int rowHalf = lane >> 4;             // 0: rows 0-7, 1: rows 8-15
  const int arow = lane & 15;                // A-matrix row for this lane

  __shared__ float    xbuf[2][16][Fn];       // TDM destination (fp32 X tiles)
  __shared__ _Float16 hbuf[16][Hn];          // staged h (f16) for MLP GEMM
  __shared__ float    partial[8][16];
  __shared__ float    lwbuf[16];

  // persistent B fragments (K = 128 -> 4 frags of 16x16x32)
  v16h Blin[4], Bw1[4];
  for (int kf = 0; kf < 4; ++kf) {
    Blin[kf] = *(const v16h*)(linPack + ((w * 4 + kf) * 512 + lane * 16));
    Bw1[kf]  = *(const v16h*)(w1Pack  + ((w * 4 + kf) * 512 + lane * 16));
  }
  const float lb   = lin_b[col];
  const float w2r  = w2[col];
  const float b2v  = b2p[0];
  const float l2w0 = l2w[0], l2w1 = l2w[1], l2w2 = l2w[2], l2bb = l2b[0];

  // C-start for MLP GEMM: g_target tile (includes mlp_b1), reused every m
  v8f creg;
  for (int v = 0; v < 8; ++v)
    creg[v] = gt[((b * Tn) + t0 + v + 8 * rowHalf) * Hn + col];

  v8f acc = {};   // h_agg partial accumulator (16 t-rows x 16 h-cols per wave)

  const int mbase = mchunk * 64;
  if (w == 0)
    issue_tdm_tile(X + (((size_t)b * Mn + mbase) * Tn + t0) * Fn, &xbuf[0][0][0]);

  for (int mi = 0; mi < 64; ++mi) {
    const int m = mbase + mi;
    const int cur = mi & 1;
    if (w == 0) wait_tdm();
    __syncthreads();
    if (w == 0 && mi + 1 < 64)
      issue_tdm_tile(X + (((size_t)b * Mn + m + 1) * Tn + t0) * Fn,
                     &xbuf[cur ^ 1][0][0]);

    // ---- h = X . lin_w^T (WMMA, f32 -> f16 convert from LDS) ----
    const float (*xb)[Fn] = xbuf[cur];
    v16h afr[4];
    for (int kf = 0; kf < 4; ++kf) {
      v16h a;
      for (int i = 0; i < 16; ++i) {
        int k = kf * 32 + ((i < 8) ? i : i + 8) + 8 * rowHalf;
        a[i] = (_Float16)xb[arow][k];
      }
      afr[kf] = a;
    }
    v8f hacc = {};
    for (int kf = 0; kf < 4; ++kf)
      hacc = __builtin_amdgcn_wmma_f32_16x16x32_f16(false, afr[kf], false, Blin[kf],
                                                    (short)0, hacc, false, false);
    float hf[8];
    for (int v = 0; v < 8; ++v) hf[v] = hacc[v] + lb;
    for (int v = 0; v < 8; ++v) hbuf[v + 8 * rowHalf][col] = (_Float16)hf[v];
    __syncthreads();

    // ---- mlp pre-act = g_target + h . W1b^T (WMMA from staged f16 h) ----
    v8f macc = creg;
    for (int kf = 0; kf < 4; ++kf) {
      v16h a;
      for (int i = 0; i < 16; ++i) {
        int k = kf * 32 + ((i < 8) ? i : i + 8) + 8 * rowHalf;
        a[i] = hbuf[arow][k];
      }
      macc = __builtin_amdgcn_wmma_f32_16x16x32_f16(false, a, false, Bw1[kf],
                                                    (short)0, macc, false, false);
    }

    // ---- w_msg row reduction: relu . w2, shuffle-reduce across 16 lanes ----
    float sv[8];
    for (int v = 0; v < 8; ++v) {
      float p = macc[v] > 0.0f ? macc[v] : 0.0f;
      sv[v] = p * w2r;
    }
    for (int off = 1; off < 16; off <<= 1)
      for (int v = 0; v < 8; ++v) sv[v] += __shfl_xor(sv[v], off, 32);
    if ((lane & 15) == 0)
      for (int v = 0; v < 8; ++v) partial[w][v + 8 * rowHalf] = sv[v];
    __syncthreads();

    // ---- learned_w per t-row ----
    if (w == 0 && lane < 16) {
      float s = 0.0f;
      for (int ww = 0; ww < 8; ++ww) s += partial[ww][lane];
      float wmsg = s + b2v;
      float cval = Cc[((size_t)b * Mn + m) * Tn + t0 + lane];
      float dm   = dist[m];
      lwbuf[lane] = l2w0 * cval + l2w1 * dm + l2w2 * wmsg + l2bb;
    }
    __syncthreads();

    for (int v = 0; v < 8; ++v) acc[v] += hf[v] * lwbuf[v + 8 * rowHalf];
    __syncthreads();
  }

  for (int v = 0; v < 8; ++v)
    atomicAdd(&hagg[((b * Tn) + t0 + v + 8 * rowHalf) * Hn + col], acc[v]);
}

// -------------------------------------------------------------------------
// 2-layer GRU, sequential over T. One 768-thread block (24 waves): wave w
// owns gate columns [16w,16w+16) of 3H=384. Separate i/h accumulators so
// n-gate computes i_n + r * h_n. h_prev lives in LDS.
// -------------------------------------------------------------------------
__global__ __launch_bounds__(768)
void gru_kernel(const float* __restrict__ hagg, const _Float16* __restrict__ gruPack,
                const float* __restrict__ bih0, const float* __restrict__ bhh0,
                const float* __restrict__ bih1, const float* __restrict__ bhh1,
                float* __restrict__ out0, float* __restrict__ dout) {
  const int tid  = threadIdx.x;
  const int w    = tid >> 5;
  const int lane = tid & 31;
  const int ncol = (w << 4) + (lane & 15);   // 0..383 gate column
  const int rowHalf = lane >> 4;
  const int arow = lane & 15;

  __shared__ _Float16 xa[16][Hn];     // x_t as f16 A-matrix (rows = batch)
  __shared__ _Float16 ha[16][Hn];     // h_prev as f16 A-matrix
  __shared__ float    hprev[16][Hn];
  __shared__ float    rbuf[16][Hn];
  __shared__ float    zbuf[16][Hn];

  const int MATH = 24 * 4 * 512;      // halves per packed [384x128] matrix

  for (int layer = 0; layer < 2; ++layer) {
    const _Float16* wihP = gruPack + (size_t)(2 * layer)     * MATH;
    const _Float16* whhP = gruPack + (size_t)(2 * layer + 1) * MATH;
    v16h Bi[4], Bh[4];
    for (int kf = 0; kf < 4; ++kf) {
      Bi[kf] = *(const v16h*)(wihP + ((w * 4 + kf) * 512 + lane * 16));
      Bh[kf] = *(const v16h*)(whhP + ((w * 4 + kf) * 512 + lane * 16));
    }
    const float* bih = layer ? bih1 : bih0;
    const float* bhh = layer ? bhh1 : bhh0;
    const float bi = bih[ncol], bh = bhh[ncol];
    const float* xsrc = layer ? out0 : hagg;
    float* dst = layer ? dout : out0;

    for (int idx = tid; idx < Bn * Hn; idx += 768) {
      int bb = idx >> 7, k = idx & 127;
      hprev[bb][k] = 0.0f;
      ha[bb][k] = (_Float16)0.0f;
    }
    __syncthreads();

    for (int t = 0; t < Tn; ++t) {
      for (int idx = tid; idx < Bn * Hn; idx += 768) {
        int bb = idx >> 7, k = idx & 127;
        xa[bb][k] = (_Float16)xsrc[(bb * Tn + t) * Hn + k];
      }
      __syncthreads();

      v8f gi = {}, gh = {};
      for (int kf = 0; kf < 4; ++kf) {
        v16h ax, ah;
        for (int i = 0; i < 16; ++i) {
          int k = kf * 32 + ((i < 8) ? i : i + 8) + 8 * rowHalf;
          ax[i] = xa[arow][k];
          ah[i] = ha[arow][k];
        }
        gi = __builtin_amdgcn_wmma_f32_16x16x32_f16(false, ax, false, Bi[kf],
                                                    (short)0, gi, false, false);
        gh = __builtin_amdgcn_wmma_f32_16x16x32_f16(false, ah, false, Bh[kf],
                                                    (short)0, gh, false, false);
      }

      if (w < 16) {   // r (waves 0-7) and z (waves 8-15)
        for (int v = 0; v < 8; ++v) {
          float g = gi[v] + gh[v] + bi + bh;
          float s = 1.0f / (1.0f + __expf(-g));
          int bb = v + 8 * rowHalf;
          if (w < 8) rbuf[bb][ncol] = s;
          else       zbuf[bb][ncol - Hn] = s;
        }
      }
      __syncthreads();

      if (w >= 16) {  // n-gate + state update (waves 16-23 cover h 0..127)
        int hc = ncol - 2 * Hn;
        for (int v = 0; v < 8; ++v) {
          int bb = v + 8 * rowHalf;
          float r  = rbuf[bb][hc];
          float nn = tanhf(gi[v] + bi + r * (gh[v] + bh));
          float z  = zbuf[bb][hc];
          float hn = (1.0f - z) * nn + z * hprev[bb][hc];
          hprev[bb][hc] = hn;
          dst[(bb * Tn + t) * Hn + hc] = hn;
        }
      }
      __syncthreads();
      if (w >= 16) {
        int hc = ncol - 2 * Hn;
        for (int v = 0; v < 8; ++v) {
          int bb = v + 8 * rowHalf;
          ha[bb][hc] = (_Float16)hprev[bb][hc];
        }
      }
      __syncthreads();
    }
    __syncthreads();
  }
}

// -------------------------------------------------------------------------
extern "C" void kernel_launch(void* const* d_in, const int* in_sizes, int n_in,
                              void* d_out, int out_size, void* d_ws, size_t ws_size,
                              hipStream_t stream) {
  const float* X     = (const float*)d_in[0];
  const float* C     = (const float*)d_in[1];
  const float* htgt  = (const float*)d_in[2];
  const float* dist  = (const float*)d_in[3];
  const float* lin_w = (const float*)d_in[4];
  const float* lin_b = (const float*)d_in[5];
  const float* w1    = (const float*)d_in[6];
  const float* b1    = (const float*)d_in[7];
  const float* w2    = (const float*)d_in[8];
  const float* b2    = (const float*)d_in[9];
  const float* l2w   = (const float*)d_in[10];
  const float* l2b   = (const float*)d_in[11];
  const float* wih0  = (const float*)d_in[12];
  const float* whh0  = (const float*)d_in[13];
  const float* bih0  = (const float*)d_in[14];
  const float* bhh0  = (const float*)d_in[15];
  const float* wih1  = (const float*)d_in[16];
  const float* whh1  = (const float*)d_in[17];
  const float* bih1  = (const float*)d_in[18];
  const float* bhh1  = (const float*)d_in[19];

  char* ws = (char*)d_ws;
  float*     hagg = (float*)(ws + 0);            // 512 KB  [B][T][H]
  float*     gt   = (float*)(ws + 524288);       // 512 KB  [B][T][H]
  float*     out0 = (float*)(ws + 1048576);      // 512 KB  [B][T][H]
  _Float16*  linP = (_Float16*)(ws + 1572864);   // 32 KB
  _Float16*  w1P  = (_Float16*)(ws + 1605632);   // 32 KB
  _Float16*  gruP = (_Float16*)(ws + 1638400);   // 384 KB (4 matrices)

  zero_kernel<<<512, 256, 0, stream>>>(hagg, Bn * Tn * Hn);
  pack_b_kernel<<<64, 256, 0, stream>>>(lin_w, Fn, 0, linP, 8 * 4 * 512);
  pack_b_kernel<<<64, 256, 0, stream>>>(w1, 2 * Hn, Hn, w1P, 8 * 4 * 512);
  pack_b_kernel<<<192, 256, 0, stream>>>(wih0, Hn, 0, gruP + 0 * 49152, 24 * 4 * 512);
  pack_b_kernel<<<192, 256, 0, stream>>>(whh0, Hn, 0, gruP + 1 * 49152, 24 * 4 * 512);
  pack_b_kernel<<<192, 256, 0, stream>>>(wih1, Hn, 0, gruP + 2 * 49152, 24 * 4 * 512);
  pack_b_kernel<<<192, 256, 0, stream>>>(whh1, Hn, 0, gruP + 3 * 49152, 24 * 4 * 512);
  gtarget_kernel<<<Bn * Tn, Hn, 0, stream>>>(htgt, w1, b1, gt);

  fused_msg_kernel<<<dim3(8, 4, 16), 256, 0, stream>>>(
      X, C, dist, linP, w1P, lin_b, w2, b2, l2w, l2b, gt, hagg);

  gru_kernel<<<1, 768, 0, stream>>>(hagg, gruP, bih0, bhh0, bih1, bhh1,
                                    out0, (float*)d_out);
}
